// DecoderLSTM_47725676593565
// MI455X (gfx1250) — matrix operand order repaired and checked by reading, not compiled, not run on hardware
//
#include <hip/hip_runtime.h>
#include <hip/hip_bf16.h>

// ---------------- problem constants ----------------
#define VV   32000
#define EE   512
#define HH   1024
#define LL   4
#define ENCC 1024
#define BB   64
#define TT   33
#define SS   64
// ---------------------------------------------------

typedef __attribute__((ext_vector_type(16))) __bf16       bf16x16;
typedef __attribute__((ext_vector_type(8)))  float        f32x8;
typedef __attribute__((ext_vector_type(4)))  unsigned int u32x4;

union FragU { bf16x16 v; u32x4 u[2]; };

// A-matrix 16x32 bf16 fragment (CDNA5 ISA 7.12.2):
// lanes 0-15 : row M=l16, K = {k0..k0+7, k0+16..k0+23}
// lanes 16-31: row M=l16, K = {k0+8..k0+15, k0+24..k0+31}
__device__ __forceinline__ bf16x16 load_frag_A(const __bf16* rowptr, int k0, int sel) {
    FragU f;
    const __bf16* p = rowptr + k0 + sel * 8;
    f.u[0] = *reinterpret_cast<const u32x4*>(p);
    f.u[1] = *reinterpret_cast<const u32x4*>(p + 16);
    return f.v;
}

// B-matrix 32x16 bf16 fragment: lane n(0-15) = col n, K = {k0..k0+7, k0+16..k0+23};
// lane n+16 = col n, K = {k0+8..k0+15, k0+24..k0+31}.  Column n of B == row n of W
// (we compute X @ W^T with W stored [N][K] row-major) -> contiguous loads.
__device__ __forceinline__ bf16x16 load_frag_B(const __bf16* rowptr, int k0, int sel) {
    FragU f;
    const __bf16* p = rowptr + k0 + sel * 8;
    f.u[0] = *reinterpret_cast<const u32x4*>(p);
    f.u[1] = *reinterpret_cast<const u32x4*>(p + 16);
    return f.v;
}

__device__ __forceinline__ f32x8 wmma_bf16(bf16x16 a, bf16x16 b, f32x8 c) {
    return __builtin_amdgcn_wmma_f32_16x16x32_bf16(
        /*neg_a=*/false, a, /*neg_b=*/false, b,
        /*c_mod=*/(short)0, c, /*reuse_a=*/false, /*reuse_b=*/false);
}

__device__ __forceinline__ float sigmoid_f(float x) {
    return 1.0f / (1.0f + __expf(-x));
}

// ---------------- fp32 -> bf16 convert (grid-stride) ----------------
__global__ void cvt_f32_bf16(const float* __restrict__ src,
                             __bf16* __restrict__ dst, size_t n) {
    size_t i = (size_t)blockIdx.x * blockDim.x + threadIdx.x;
    size_t stride = (size_t)gridDim.x * blockDim.x;
    for (; i < n; i += stride) dst[i] = (__bf16)src[i];
}

// ---------------- attention + softmax + context + embedding ----------------
// one block per batch element; h_top is previous step's top-layer hidden (bf16)
__global__ __launch_bounds__(256)
void attn_embed_kernel(const __bf16* __restrict__ h_top,     // [B,H]
                       const float*  __restrict__ enc_out,   // [B,S,ENC]
                       const unsigned char* __restrict__ enc_mask, // [B,S]
                       const int*    __restrict__ tgt,       // [B,T]
                       int t,
                       const float*  __restrict__ embedding, // [V,E]
                       const float*  __restrict__ w_attn,    // [ENC,H]
                       __bf16* __restrict__ inp0,            // [B, E+ENC]
                       __bf16* __restrict__ projin) {        // [B, H+ENC]
    __shared__ float h_sh[HH];
    __shared__ float wdh_sh[ENCC];
    __shared__ float red_sh[256];
    __shared__ float sc_sh[SS];

    const int b = blockIdx.x;
    const int tid = threadIdx.x;

    for (int j = tid; j < HH; j += 256) h_sh[j] = (float)h_top[(size_t)b * HH + j];
    __syncthreads();

    // wdh = h_top @ w_attn^T   (wdh[j] = dot(w_attn[j,:], h))
    for (int j = tid; j < ENCC; j += 256) {
        const float* wr = w_attn + (size_t)j * HH;
        float s = 0.f;
        for (int k = 0; k < HH; ++k) s = fmaf(wr[k], h_sh[k], s);
        wdh_sh[j] = s;
    }
    __syncthreads();

    // scores[s] = dot(enc_out[b,s,:], wdh): 4 threads per s
    {
        const int sidx = tid >> 2, q = tid & 3;
        const float* er = enc_out + ((size_t)b * SS + sidx) * ENCC + q * 256;
        const float* wr = wdh_sh + q * 256;
        float p = 0.f;
        for (int k = 0; k < 256; ++k) p = fmaf(er[k], wr[k], p);
        red_sh[tid] = p;
    }
    __syncthreads();
    if (tid < SS) {
        float v = red_sh[4 * tid] + red_sh[4 * tid + 1] +
                  red_sh[4 * tid + 2] + red_sh[4 * tid + 3];
        if (enc_mask[(size_t)b * SS + tid]) v = -1e30f;
        sc_sh[tid] = v;
    }
    __syncthreads();

    // softmax stats (redundant per-thread; 64 elems, trivial)
    float mx = -1e30f;
    for (int s = 0; s < SS; ++s) mx = fmaxf(mx, sc_sh[s]);
    float den = 0.f;
    for (int s = 0; s < SS; ++s) den += __expf(sc_sh[s] - mx);
    const float inv = 1.0f / den;

    // ctx[e] = sum_s softmax(s) * enc_out[b,s,e]
    for (int e = tid; e < ENCC; e += 256) {
        float cv = 0.f;
        for (int s = 0; s < SS; ++s)
            cv = fmaf(__expf(sc_sh[s] - mx),
                      enc_out[((size_t)b * SS + s) * ENCC + e], cv);
        cv *= inv;
        inp0[(size_t)b * (EE + ENCC) + EE + e]   = (__bf16)cv;
        projin[(size_t)b * (HH + ENCC) + HH + e] = (__bf16)cv;
    }

    // x = embedding[tgt[b, t]]
    const int y = tgt[b * TT + t];
    for (int e = tid; e < EE; e += 256)
        inp0[(size_t)b * (EE + ENCC) + e] = (__bf16)embedding[(size_t)y * EE + e];
}

// ---------------- LSTM gate pre-activation GEMM ----------------
// g[64, 4H] = inp @ w_ih^T + h_in @ w_hh^T   (fp32 out, bias added later)
// grid = 4H/64 = 64 blocks, 128 threads (4 waves). Wave owns one 16-wide
// column strip of the 4096-wide gate output and all 4 M-tiles (acc = 32 VGPRs
// -> real occupancy; weights still read exactly once per step).
__global__ __launch_bounds__(128)
void lstm_gemm_kernel(const __bf16* __restrict__ inp,  int Kin,   // [B,Kin]
                      const __bf16* __restrict__ w_ih,            // [4H,Kin]
                      const __bf16* __restrict__ h_in,            // [B,H]
                      const __bf16* __restrict__ w_hh,            // [4H,H]
                      float* __restrict__ gbuf) {                 // [B,4H]
    const int tid  = threadIdx.x;
    const int wv   = tid >> 5;
    const int lane = tid & 31;
    const int sel  = lane >> 4;
    const int l16  = lane & 15;
    const int ncg  = blockIdx.x * 64 + wv * 16 + l16;   // gate-output column (0..4095)

    f32x8 acc[4] = {};

    // phase 1: inp @ w_ih^T
    {
        const __bf16* arow = inp + (size_t)l16 * Kin;
        const __bf16* brow = w_ih + (size_t)ncg * Kin;
        for (int k0 = 0; k0 < Kin; k0 += 32) {
            __builtin_prefetch(brow + k0 + 512, 0, 0);   // global_prefetch_b8
            bf16x16 b = load_frag_B(brow, k0, sel);
#pragma unroll
            for (int mt = 0; mt < 4; ++mt) {
                bf16x16 a = load_frag_A(arow + (size_t)mt * 16 * Kin, k0, sel);
                acc[mt] = wmma_bf16(a, b, acc[mt]);
            }
        }
    }
    // phase 2: h_in @ w_hh^T  (K = H)
    {
        const __bf16* arow = h_in + (size_t)l16 * HH;
        const __bf16* brow = w_hh + (size_t)ncg * HH;
        for (int k0 = 0; k0 < HH; k0 += 32) {
            __builtin_prefetch(brow + k0 + 512, 0, 0);
            bf16x16 b = load_frag_B(brow, k0, sel);
#pragma unroll
            for (int mt = 0; mt < 4; ++mt) {
                bf16x16 a = load_frag_A(arow + (size_t)mt * 16 * HH, k0, sel);
                acc[mt] = wmma_bf16(a, b, acc[mt]);
            }
        }
    }

    // write pre-activations (C/D layout: VGPR r -> row mt*16 + r + 8*sel)
#pragma unroll
    for (int mt = 0; mt < 4; ++mt)
#pragma unroll
        for (int r = 0; r < 8; ++r) {
            const int m = mt * 16 + r + sel * 8;
            gbuf[(size_t)m * (4 * HH) + ncg] = acc[mt][r];
        }
}

// ---------------- LSTM elementwise gates ----------------
// grid-stride over B*H elements; PyTorch gate order i, f, g, o
__global__ __launch_bounds__(256)
void lstm_gate_kernel(const float* __restrict__ gbuf,   // [B,4H]
                      const float* __restrict__ b_ih,   // [4H]
                      const float* __restrict__ b_hh,   // [4H]
                      float*  __restrict__ c_state,     // [B,H] in-place
                      __bf16* __restrict__ h_out,       // [B,H]
                      __bf16* __restrict__ projin) {    // [B,H+ENC] or null
    const int total = BB * HH;
    for (int i = blockIdx.x * 256 + threadIdx.x; i < total; i += gridDim.x * 256) {
        const int m = i >> 10;          // batch row
        const int n = i & (HH - 1);     // hidden column
        const size_t gb = (size_t)m * (4 * HH);
        const float iv = sigmoid_f(gbuf[gb + n]            + b_ih[n]            + b_hh[n]);
        const float fv = sigmoid_f(gbuf[gb + HH + n]       + b_ih[HH + n]       + b_hh[HH + n]);
        const float gv = tanhf(    gbuf[gb + 2 * HH + n]   + b_ih[2 * HH + n]   + b_hh[2 * HH + n]);
        const float ov = sigmoid_f(gbuf[gb + 3 * HH + n]   + b_ih[3 * HH + n]   + b_hh[3 * HH + n]);
        const float cn = fv * c_state[i] + iv * gv;
        const float hn = ov * tanhf(cn);
        c_state[i] = cn;
        h_out[i]   = (__bf16)hn;
        if (projin) projin[(size_t)m * (HH + ENCC) + n] = (__bf16)hn;
    }
}

// ---------------- vocab projection GEMM ----------------
// grid = V/64 blocks, 128 threads; wave owns a 16-wide vocab strip, 4 M-tiles.
__global__ __launch_bounds__(128)
void proj_kernel(const __bf16* __restrict__ projin,   // [B, H+ENC]
                 const __bf16* __restrict__ w_proj,   // [V, H+ENC]
                 const float*  __restrict__ b_proj,   // [V]
                 float* __restrict__ out,             // [B, T-1, V]
                 int t) {
    const int tid  = threadIdx.x;
    const int wv   = tid >> 5;
    const int lane = tid & 31;
    const int sel  = lane >> 4;
    const int l16  = lane & 15;
    const int ncg  = blockIdx.x * 64 + wv * 16 + l16;   // vocab column
    const int K    = HH + ENCC;                          // 2048

    f32x8 acc[4] = {};
    const __bf16* arow = projin + (size_t)l16 * K;
    const __bf16* brow = w_proj + (size_t)ncg * K;

    for (int k0 = 0; k0 < K; k0 += 32) {
        __builtin_prefetch(brow + k0 + 512, 0, 0);       // global_prefetch_b8
        bf16x16 b = load_frag_B(brow, k0, sel);
#pragma unroll
        for (int mt = 0; mt < 4; ++mt) {
            bf16x16 a = load_frag_A(arow + (size_t)mt * 16 * K, k0, sel);
            acc[mt] = wmma_bf16(a, b, acc[mt]);
        }
    }

    const float bias = b_proj[ncg];
#pragma unroll
    for (int mt = 0; mt < 4; ++mt)
#pragma unroll
        for (int r = 0; r < 8; ++r) {
            const int bRow = mt * 16 + r + sel * 8;      // batch index
            out[((size_t)bRow * (TT - 1) + t) * VV + ncg] = acc[mt][r] + bias;
        }
}

// ---------------- host orchestration ----------------
extern "C" void kernel_launch(void* const* d_in, const int* in_sizes, int n_in,
                              void* d_out, int out_size, void* d_ws, size_t ws_size,
                              hipStream_t stream) {
    (void)in_sizes; (void)n_in; (void)out_size; (void)ws_size;

    const int*           tgt      = (const int*)d_in[0];
    const float*         h0       = (const float*)d_in[1];
    const float*         c0       = (const float*)d_in[2];
    const float*         enc_out  = (const float*)d_in[3];
    const unsigned char* enc_mask = (const unsigned char*)d_in[4];
    const float*         emb      = (const float*)d_in[5];
    const float*         w_attn   = (const float*)d_in[6];
    const float*         w_ih0    = (const float*)d_in[7];
    const float*         w_ihr    = (const float*)d_in[8];
    const float*         w_hh     = (const float*)d_in[9];
    const float*         b_ih     = (const float*)d_in[10];
    const float*         b_hh     = (const float*)d_in[11];
    const float*         w_proj   = (const float*)d_in[12];
    const float*         b_proj   = (const float*)d_in[13];
    float* out = (float*)d_out;

    // ---- workspace carve-up (256B aligned) ----
    char* base = (char*)d_ws;
    size_t off = 0;
    auto carve = [&](size_t bytes) -> char* {
        char* p = base + off;
        off = (off + bytes + 255) & ~(size_t)255;
        return p;
    };
    const size_t n_wproj = (size_t)VV * (HH + ENCC);
    const size_t n_wih0  = (size_t)4 * HH * (EE + ENCC);
    const size_t n_wihr  = (size_t)(LL - 1) * 4 * HH * HH;
    const size_t n_whh   = (size_t)LL * 4 * HH * HH;
    const size_t n_state = (size_t)LL * BB * HH;

    __bf16* wsProj = (__bf16*)carve(n_wproj * 2);
    __bf16* wsIh0  = (__bf16*)carve(n_wih0 * 2);
    __bf16* wsIhr  = (__bf16*)carve(n_wihr * 2);
    __bf16* wsHh   = (__bf16*)carve(n_whh * 2);
    __bf16* hA     = (__bf16*)carve(n_state * 2);
    __bf16* hB     = (__bf16*)carve(n_state * 2);
    float*  cS     = (float*) carve(n_state * 4);
    __bf16* inp0   = (__bf16*)carve((size_t)BB * (EE + ENCC) * 2);
    __bf16* projin = (__bf16*)carve((size_t)BB * (HH + ENCC) * 2);
    float*  gbuf   = (float*) carve((size_t)BB * 4 * HH * 4);

    // ---- one-time (per call) weight conversion to bf16 ----
    cvt_f32_bf16<<<2048, 256, 0, stream>>>(w_proj, wsProj, n_wproj);
    cvt_f32_bf16<<<1024, 256, 0, stream>>>(w_ih0,  wsIh0,  n_wih0);
    cvt_f32_bf16<<<1024, 256, 0, stream>>>(w_ihr,  wsIhr,  n_wihr);
    cvt_f32_bf16<<<1024, 256, 0, stream>>>(w_hh,   wsHh,   n_whh);
    cvt_f32_bf16<<<256,  256, 0, stream>>>(h0,     hA,     n_state);
    hipMemcpyAsync(cS, c0, n_state * 4, hipMemcpyDeviceToDevice, stream);

    const size_t layH  = (size_t)BB * HH;       // per-layer state stride (elems)
    const size_t layW  = (size_t)4 * HH * HH;   // per-layer weight stride
    const size_t layB  = (size_t)4 * HH;        // per-layer bias stride

    for (int t = 0; t < TT - 1; ++t) {
        __bf16* hin  = (t & 1) ? hB : hA;
        __bf16* hout = (t & 1) ? hA : hB;

        attn_embed_kernel<<<BB, 256, 0, stream>>>(
            hin + (size_t)(LL - 1) * layH, enc_out, enc_mask, tgt, t,
            emb, w_attn, inp0, projin);

        for (int l = 0; l < LL; ++l) {
            const __bf16* lin  = (l == 0) ? inp0 : hout + (size_t)(l - 1) * layH;
            const int     kin  = (l == 0) ? (EE + ENCC) : HH;
            const __bf16* wih  = (l == 0) ? wsIh0 : wsIhr + (size_t)(l - 1) * layW;

            lstm_gemm_kernel<<<4 * HH / 64, 128, 0, stream>>>(
                lin, kin, wih,
                hin + (size_t)l * layH,
                wsHh + (size_t)l * layW,
                gbuf);

            lstm_gate_kernel<<<64, 256, 0, stream>>>(
                gbuf,
                b_ih + (size_t)l * layB,
                b_hh + (size_t)l * layB,
                cS + (size_t)l * layH,
                hout + (size_t)l * layH,
                (l == LL - 1) ? projin : (__bf16*)nullptr);
        }

        // logits = [h_top|ctx] @ w_proj^T + b_proj
        proj_kernel<<<VV / 64, 128, 0, stream>>>(projin, wsProj, b_proj, out, t);
    }
}
